// RDQuantLinear_36610301231405
// MI455X (gfx1250) — compile-verified, matrix operand order checked by reading, not run on hardware
//
#include <hip/hip_runtime.h>
#include <hip/hip_bf16.h>

// ---------------------------------------------------------------------------
// Mixed-quantization linear (NVFP4 + FP8 + FP16 groups) on gfx1250.
// Two-phase: (1) one-time dequant/convert of x and W to bf16 in workspace,
// (2) double-buffered bf16 WMMA GEMM with async global->LDS staging
//     (ASYNCcnt path) and permuted-scatter + bias epilogue.
// Falls back to a fully fused dequant-in-GEMM kernel if workspace is small.
// ---------------------------------------------------------------------------

typedef __attribute__((ext_vector_type(16))) __bf16 v16bf;
typedef __attribute__((ext_vector_type(8)))  float  v8f;

// 128-bit vector pointee type expected by the async-load builtin
typedef int v4i __attribute__((vector_size(16)));
typedef __attribute__((address_space(1))) v4i glb_v4i;
typedef __attribute__((address_space(3))) v4i lds_v4i;

#define MDIM 4096
#define KDIM 4096
#define NDIM 4096
#define N_NVFP4 2048
#define N_FP8   1024
#define BLOCKQ  16

#define BM 128
#define BN 128
#define BK 64
#define ASTR (BK + 8)   // 72 halves per row -> 144B row stride, 16B-aligned
#define NTHREADS 256

#if defined(__has_builtin)
#if __has_builtin(__builtin_amdgcn_global_load_async_to_lds_b128)
#define HAVE_ASYNC_LDS 1
#endif
#if __has_builtin(__builtin_amdgcn_s_wait_asynccnt)
#define HAVE_WAIT_ASYNC_BUILTIN 1
#endif
#endif

__device__ __forceinline__ void async_wait0() {
#if defined(HAVE_ASYNC_LDS)
#if defined(HAVE_WAIT_ASYNC_BUILTIN)
    __builtin_amdgcn_s_wait_asynccnt(0);
#else
    asm volatile("s_wait_asynccnt 0x0" ::: "memory");
#endif
#endif
}

// stage one 16B chunk: global (bf16) -> LDS, async if the toolchain has it
__device__ __forceinline__ void stage_copy16(const unsigned short* __restrict__ gsrc,
                                             unsigned short* ldst) {
#if defined(HAVE_ASYNC_LDS)
    __builtin_amdgcn_global_load_async_to_lds_b128(
        (glb_v4i*)gsrc, (lds_v4i*)ldst, 0, 0);
#else
    *(uint4*)ldst = *(const uint4*)gsrc;
#endif
}

// round-to-nearest-even float -> bf16 (as u16 in low bits)
__device__ __forceinline__ unsigned bf16_rne(float f) {
    unsigned u = __float_as_uint(f);
    return ((u + 0x7fffu + ((u >> 16) & 1u)) >> 16) & 0xffffu;
}
__device__ __forceinline__ unsigned pack_bf16x2(float a, float b) {
    return bf16_rne(a) | (bf16_rne(b) << 16);
}

// NVFP4 (e2m1) decode from 4-bit code, built directly as fp32 bits.
// mag(m): 0,0.5,1,1.5,2,3,4,6 for m=0..7 ; sign = bit3.
__device__ __forceinline__ float nvfp4_dec(int c) {
    int m = c & 7;
    unsigned bits = (m == 0) ? 0u
        : (((126u + (unsigned)(m >> 1)) << 23) |
           ((m >= 2) ? (((unsigned)(m & 1)) << 22) : 0u));
    bits |= ((unsigned)(c & 8)) << 28;
    return __uint_as_float(bits);
}

// Invert the permutation: fwd[inv_perm[j]] = j
__global__ void build_fwd_perm(const long long* __restrict__ inv_perm,
                               int* __restrict__ fwd, int n) {
    int j = blockIdx.x * blockDim.x + threadIdx.x;
    if (j < n) fwd[(int)inv_perm[j]] = j;
}

// ---------------------------------------------------------------------------
// Phase 1a: x fp32 -> bf16 (8 elements / thread)
// ---------------------------------------------------------------------------
__global__ __launch_bounds__(NTHREADS)
void convert_x_bf16(const float* __restrict__ x, unsigned short* __restrict__ xb) {
    const size_t gid = (size_t)blockIdx.x * NTHREADS + threadIdx.x;
    const float4* p = (const float4*)x + gid * 2;
    const float4 v0 = p[0];
    const float4 v1 = p[1];
    uint4 o;
    o.x = pack_bf16x2(v0.x, v0.y);
    o.y = pack_bf16x2(v0.z, v0.w);
    o.z = pack_bf16x2(v1.x, v1.y);
    o.w = pack_bf16x2(v1.z, v1.w);
    ((uint4*)xb)[gid] = o;
}

// ---------------------------------------------------------------------------
// Phase 1b: dequantize all three weight groups -> bf16 (8 elements / thread)
// ---------------------------------------------------------------------------
__global__ __launch_bounds__(NTHREADS)
void dequant_w_bf16(const int*   __restrict__ idx4,
                    const float* __restrict__ sc4,
                    const float* __restrict__ gscale_p,
                    const float* __restrict__ wfp8,
                    const float* __restrict__ fp8s_p,
                    const float* __restrict__ wfp16,
                    unsigned short* __restrict__ wb) {
    const size_t gid = (size_t)blockIdx.x * NTHREADS + threadIdx.x;
    const int row = (int)(gid >> 9);        // 512 chunks of 8 per 4096-wide row
    const int k   = ((int)gid & 511) * 8;   // 8 elements all in one 16-block
    uint4 o;
    if (row < N_NVFP4) {
        const int4* cp = (const int4*)(idx4 + (size_t)row * KDIM + k);
        const int4 c0 = cp[0];
        const int4 c1 = cp[1];
        const float s = sc4[(size_t)row * (KDIM / BLOCKQ) + (k >> 4)] * gscale_p[0];
        o.x = pack_bf16x2(nvfp4_dec(c0.x) * s, nvfp4_dec(c0.y) * s);
        o.y = pack_bf16x2(nvfp4_dec(c0.z) * s, nvfp4_dec(c0.w) * s);
        o.z = pack_bf16x2(nvfp4_dec(c1.x) * s, nvfp4_dec(c1.y) * s);
        o.w = pack_bf16x2(nvfp4_dec(c1.z) * s, nvfp4_dec(c1.w) * s);
    } else if (row < N_NVFP4 + N_FP8) {
        const float4* p = (const float4*)(wfp8 + (size_t)(row - N_NVFP4) * KDIM + k);
        const float s = fp8s_p[0];
        const float4 v0 = p[0];
        const float4 v1 = p[1];
        o.x = pack_bf16x2(v0.x * s, v0.y * s);
        o.y = pack_bf16x2(v0.z * s, v0.w * s);
        o.z = pack_bf16x2(v1.x * s, v1.y * s);
        o.w = pack_bf16x2(v1.z * s, v1.w * s);
    } else {
        const float4* p = (const float4*)(wfp16 + (size_t)(row - N_NVFP4 - N_FP8) * KDIM + k);
        const float4 v0 = p[0];
        const float4 v1 = p[1];
        o.x = pack_bf16x2(v0.x, v0.y);
        o.y = pack_bf16x2(v0.z, v0.w);
        o.z = pack_bf16x2(v1.x, v1.y);
        o.w = pack_bf16x2(v1.z, v1.w);
    }
    ((uint4*)wb)[gid] = o;
}

// ---------------------------------------------------------------------------
// Phase 2: pure bf16 WMMA GEMM, double-buffered LDS, async staging.
// 128x128 tile per 8-wave block, BK=64; each wave owns a 64x32 sub-tile
// (4x2 accumulators of v_wmma_f32_16x16x32_bf16).
// ---------------------------------------------------------------------------
__global__ __launch_bounds__(NTHREADS)
void gemm_bf16_wmma(const unsigned short* __restrict__ xb,
                    const unsigned short* __restrict__ wb,
                    const float* __restrict__ bias,
                    const int*   __restrict__ fwd,
                    float*       __restrict__ out) {
    __shared__ unsigned short As[2][BM * ASTR];
    __shared__ unsigned short Bs[2][BN * ASTR];

    const int tid = threadIdx.x;
    const int n0  = blockIdx.x * BN;
    const int m0  = blockIdx.y * BM;

    const int wave = tid >> 5;
    const int lane = tid & 31;
    const int wm = (wave >> 2) * 64;
    const int wn = (wave & 3) * 32;
    const int lh = lane >> 4;
    const int lr = lane & 15;

    // cooperative copy coords: 8 uint4 (128B) per 64-bf16 row, 32 rows / pass
    const int cu = tid & 7;    // uint4 column within row
    const int ru = tid >> 3;   // starting row, advances by 32

    v8f acc[4][2];
#pragma unroll
    for (int i = 0; i < 4; i++)
#pragma unroll
        for (int j = 0; j < 2; j++)
#pragma unroll
            for (int e = 0; e < 8; e++) acc[i][j][e] = 0.0f;

    // issue the first stage into buffer 0
#pragma unroll
    for (int i = 0; i < 4; i++) {
        const int row = ru + i * 32;
        stage_copy16(xb + (size_t)(m0 + row) * KDIM + cu * 8,
                     &As[0][row * ASTR + cu * 8]);
        stage_copy16(wb + (size_t)(n0 + row) * KDIM + cu * 8,
                     &Bs[0][row * ASTR + cu * 8]);
    }
    async_wait0();
    __syncthreads();

    int cur = 0;
    for (int k0 = 0; k0 < KDIM; k0 += BK) {
        // prefetch stage k0+BK into the other buffer while computing this one
        if (k0 + BK < KDIM) {
            const int nxt = cur ^ 1;
#pragma unroll
            for (int i = 0; i < 4; i++) {
                const int row = ru + i * 32;
                stage_copy16(xb + (size_t)(m0 + row) * KDIM + (k0 + BK) + cu * 8,
                             &As[nxt][row * ASTR + cu * 8]);
                stage_copy16(wb + (size_t)(n0 + row) * KDIM + (k0 + BK) + cu * 8,
                             &Bs[nxt][row * ASTR + cu * 8]);
            }
        }

#pragma unroll
        for (int ks = 0; ks < 2; ks++) {
            const int kb = ks * 32 + lh * 8;  // per-lane K base (bf16 frag layout)
            v16bf a[4], b[2];
#pragma unroll
            for (int mt = 0; mt < 4; mt++) {
                const unsigned short* p = &As[cur][(wm + mt * 16 + lr) * ASTR + kb];
                union { v16bf v; uint4 q[2]; } u;
                u.q[0] = *(const uint4*)p;          // K = kb .. kb+7
                u.q[1] = *(const uint4*)(p + 16);   // K = kb+16 .. kb+23
                a[mt] = u.v;
            }
#pragma unroll
            for (int nt = 0; nt < 2; nt++) {
                const unsigned short* p = &Bs[cur][(wn + nt * 16 + lr) * ASTR + kb];
                union { v16bf v; uint4 q[2]; } u;
                u.q[0] = *(const uint4*)p;
                u.q[1] = *(const uint4*)(p + 16);
                b[nt] = u.v;
            }
#pragma unroll
            for (int mt = 0; mt < 4; mt++)
#pragma unroll
                for (int nt = 0; nt < 2; nt++)
                    acc[mt][nt] = __builtin_amdgcn_wmma_f32_16x16x32_bf16(
                        false, a[mt], false, b[nt],
                        (short)0, acc[mt][nt], false, false);
        }

        async_wait0();       // next-stage DMA landed
        __syncthreads();     // everyone done reading cur / writing nxt
        cur ^= 1;
    }

#pragma unroll
    for (int mt = 0; mt < 4; mt++) {
#pragma unroll
        for (int nt = 0; nt < 2; nt++) {
            const int gn = n0 + wn + nt * 16 + lr;
            const int j  = fwd[gn];
            const float bj = bias[j];
#pragma unroll
            for (int r = 0; r < 8; r++) {
                const int gm = m0 + wm + mt * 16 + lh * 8 + r;
                out[(size_t)gm * NDIM + j] = acc[mt][nt][r] + bj;
            }
        }
    }
}

// ---------------------------------------------------------------------------
// Fallback: fully fused dequant-in-GEMM (used only if workspace is too small
// for the bf16 staging buffers).
// ---------------------------------------------------------------------------
__global__ __launch_bounds__(NTHREADS)
void fused_qlinear_wmma(const float* __restrict__ x,
                        const int*   __restrict__ idx4,
                        const float* __restrict__ sc4,
                        const float* __restrict__ gscale_p,
                        const float* __restrict__ wfp8,
                        const float* __restrict__ fp8s_p,
                        const float* __restrict__ wfp16,
                        const float* __restrict__ bias,
                        const int*   __restrict__ fwd,
                        float*       __restrict__ out) {
    __shared__ unsigned short As[BM * ASTR];
    __shared__ unsigned short Bs[BN * ASTR];

    const int tid = threadIdx.x;
    const int n0  = blockIdx.x * BN;
    const int m0  = blockIdx.y * BM;
    const int grp = (n0 < N_NVFP4) ? 0 : ((n0 < N_NVFP4 + N_FP8) ? 1 : 2);

    const float gscale = gscale_p[0];
    const float fp8s   = fp8s_p[0];

    const int wave = tid >> 5;
    const int lane = tid & 31;
    const int wm = (wave >> 2) * 64;
    const int wn = (wave & 3) * 32;
    const int lh = lane >> 4;
    const int lr = lane & 15;

    const int c4 = tid & 15;
    const int r0 = tid >> 4;

    v8f acc[4][2];
#pragma unroll
    for (int i = 0; i < 4; i++)
#pragma unroll
        for (int j = 0; j < 2; j++)
#pragma unroll
            for (int e = 0; e < 8; e++) acc[i][j][e] = 0.0f;

    for (int k0 = 0; k0 < KDIM; k0 += BK) {
#pragma unroll
        for (int i = 0; i < 8; i++) {
            const int row = r0 + i * 16;
            const float4 v = *(const float4*)(x + (size_t)(m0 + row) * KDIM + k0 + c4 * 4);
            *(uint2*)(&As[row * ASTR + c4 * 4]) =
                make_uint2(pack_bf16x2(v.x, v.y), pack_bf16x2(v.z, v.w));
        }
        if (grp == 0) {
#pragma unroll
            for (int i = 0; i < 8; i++) {
                const int row  = r0 + i * 16;
                const int grow = n0 + row;
                const int4 c = *(const int4*)(idx4 + (size_t)grow * KDIM + k0 + c4 * 4);
                const float s = sc4[(size_t)grow * (KDIM / BLOCKQ) + (k0 + c4 * 4) / BLOCKQ] * gscale;
                *(uint2*)(&Bs[row * ASTR + c4 * 4]) =
                    make_uint2(pack_bf16x2(nvfp4_dec(c.x) * s, nvfp4_dec(c.y) * s),
                               pack_bf16x2(nvfp4_dec(c.z) * s, nvfp4_dec(c.w) * s));
            }
        } else if (grp == 1) {
#pragma unroll
            for (int i = 0; i < 8; i++) {
                const int row = r0 + i * 16;
                const int rl  = n0 + row - N_NVFP4;
                const float4 v = *(const float4*)(wfp8 + (size_t)rl * KDIM + k0 + c4 * 4);
                *(uint2*)(&Bs[row * ASTR + c4 * 4]) =
                    make_uint2(pack_bf16x2(v.x * fp8s, v.y * fp8s),
                               pack_bf16x2(v.z * fp8s, v.w * fp8s));
            }
        } else {
#pragma unroll
            for (int i = 0; i < 8; i++) {
                const int row = r0 + i * 16;
                const int rl  = n0 + row - (N_NVFP4 + N_FP8);
                const float4 v = *(const float4*)(wfp16 + (size_t)rl * KDIM + k0 + c4 * 4);
                *(uint2*)(&Bs[row * ASTR + c4 * 4]) =
                    make_uint2(pack_bf16x2(v.x, v.y), pack_bf16x2(v.z, v.w));
            }
        }
        __syncthreads();

#pragma unroll
        for (int ks = 0; ks < 2; ks++) {
            const int kb = ks * 32 + lh * 8;
            v16bf a[4], b[2];
#pragma unroll
            for (int mt = 0; mt < 4; mt++) {
                const unsigned short* p = &As[(wm + mt * 16 + lr) * ASTR + kb];
                union { v16bf v; uint4 q[2]; } u;
                u.q[0] = *(const uint4*)p;
                u.q[1] = *(const uint4*)(p + 16);
                a[mt] = u.v;
            }
#pragma unroll
            for (int nt = 0; nt < 2; nt++) {
                const unsigned short* p = &Bs[(wn + nt * 16 + lr) * ASTR + kb];
                union { v16bf v; uint4 q[2]; } u;
                u.q[0] = *(const uint4*)p;
                u.q[1] = *(const uint4*)(p + 16);
                b[nt] = u.v;
            }
#pragma unroll
            for (int mt = 0; mt < 4; mt++)
#pragma unroll
                for (int nt = 0; nt < 2; nt++)
                    acc[mt][nt] = __builtin_amdgcn_wmma_f32_16x16x32_bf16(
                        false, a[mt], false, b[nt],
                        (short)0, acc[mt][nt], false, false);
        }
        __syncthreads();
    }

#pragma unroll
    for (int mt = 0; mt < 4; mt++) {
#pragma unroll
        for (int nt = 0; nt < 2; nt++) {
            const int gn = n0 + wn + nt * 16 + lr;
            const int j  = fwd[gn];
            const float bj = bias[j];
#pragma unroll
            for (int r = 0; r < 8; r++) {
                const int gm = m0 + wm + mt * 16 + lh * 8 + r;
                out[(size_t)gm * NDIM + j] = acc[mt][nt][r] + bj;
            }
        }
    }
}

extern "C" void kernel_launch(void* const* d_in, const int* in_sizes, int n_in,
                              void* d_out, int out_size, void* d_ws, size_t ws_size,
                              hipStream_t stream) {
    const float*     x     = (const float*)d_in[0];
    const int*       idx4  = (const int*)d_in[1];
    const float*     sc4   = (const float*)d_in[2];
    const float*     gs    = (const float*)d_in[3];
    const float*     wfp8  = (const float*)d_in[4];
    const float*     f8s   = (const float*)d_in[5];
    const float*     wfp16 = (const float*)d_in[6];
    const float*     bias  = (const float*)d_in[7];
    const long long* invp  = (const long long*)d_in[8];
    float*           out   = (float*)d_out;

    int* fwd = (int*)d_ws;  // 4096 ints at the front of the workspace
    build_fwd_perm<<<NDIM / 256, 256, 0, stream>>>(invp, fwd, NDIM);

    const size_t PERM_BYTES = 16384;
    const size_t XB_BYTES = (size_t)MDIM * KDIM * sizeof(unsigned short);
    const size_t WB_BYTES = (size_t)NDIM * KDIM * sizeof(unsigned short);
    dim3 grid(NDIM / BN, MDIM / BM);

    if (ws_size >= PERM_BYTES + XB_BYTES + WB_BYTES) {
        unsigned short* xb = (unsigned short*)((char*)d_ws + PERM_BYTES);
        unsigned short* wb = xb + (size_t)MDIM * KDIM;
        convert_x_bf16<<<(MDIM * (size_t)KDIM / 8) / NTHREADS, NTHREADS, 0, stream>>>(x, xb);
        dequant_w_bf16<<<(NDIM * (size_t)KDIM / 8) / NTHREADS, NTHREADS, 0, stream>>>(
            idx4, sc4, gs, wfp8, f8s, wfp16, wb);
        gemm_bf16_wmma<<<grid, NTHREADS, 0, stream>>>(xb, wb, bias, fwd, out);
    } else {
        fused_qlinear_wmma<<<grid, NTHREADS, 0, stream>>>(
            x, idx4, sc4, gs, wfp8, f8s, wfp16, bias, fwd, out);
    }
}
